// SpringModel_74612171866459
// MI455X (gfx1250) — matrix operand order repaired and checked by reading, not compiled, not run on hardware
//
#include <hip/hip_runtime.h>
#include <hip/hip_bf16.h>

typedef _Float16 f16;
typedef __attribute__((ext_vector_type(16))) _Float16 v16h;
typedef __attribute__((ext_vector_type(8)))  _Float16 v8h;
typedef __attribute__((ext_vector_type(8)))  float    v8f;

#define NN 100000
#define NE 1600000
#define LDST 72   // LDS row stride in halfs (pads banks: 36*m mod 64 distinct for m=0..15)

union V16u { v16h v; v8h h[2]; f16 e[16]; };

// Packed-weight offsets (in halfs) inside the f16 weight arena.
static constexpr int OFF_PE_W1 = 0;      // pos_enc.w1  [2x64 ->K32]   2048
static constexpr int OFF_PE_W2 = 2048;   // pos_enc.w2  [64x64]        4096
static constexpr int OFF_VE_W1 = 6144;   // vel_enc.w1                 2048
static constexpr int OFF_VE_W2 = 8192;   // vel_enc.w2                 4096
static constexpr int OFF_EE_W1 = 12288;  // edge_enc.w1 [4x64 ->K32]   2048
static constexpr int OFF_EE_W2 = 14336;  // edge_enc.w2                4096
static constexpr int OFF_PR_W1 = 18432;  // process.w1  [192x64]      12288
static constexpr int OFF_PR_W2 = 30720;  // process.w2                 4096
static constexpr int OFF_PD_W1 = 34816;  // pos_dec.w1                 4096
static constexpr int OFF_PD_W2 = 38912;  // pos_dec.w2  [64x2 ->N16]   1024
static constexpr int OFF_VD_W1 = 39936;  // vel_dec.w1                 4096
static constexpr int OFF_VD_W2 = 44032;  // vel_dec.w2                 1024

__device__ __forceinline__ v8f wmma32(v16h a, v16h b, v8f c) {
  return __builtin_amdgcn_wmma_f32_16x16x32_f16(false, a, false, b, (short)0, c,
                                                false, false);
}

// Accumulate a 16x64 tile: c[nb] += A(16x(32*KT)) @ Wpk(tile nb, kstep kt).
// Packed B layout: halfs[((nb*KT + kt)*32 + lane)*16 + j], k = kt*32 + 8g + (j&7) + 16*(j>>3).
template <int KT, typename AF>
__device__ __forceinline__ void mm_acc(AF&& getA, const f16* __restrict__ wpk,
                                       v8f* c, int lane) {
#pragma unroll
  for (int kt = 0; kt < KT; ++kt) {
    v16h a = getA(kt);
#pragma unroll
    for (int nb = 0; nb < 4; ++nb) {
      v16h b = *(const v16h*)(wpk + (((nb * KT + kt) * 32 + lane) << 4));
      c[nb] = wmma32(a, b, c[nb]);
    }
  }
}

// A-fragment from an LDS 16xLDST f16 tile. A layout: m = lane&15,
// j<8 -> k = kt*32 + 8g + j ; j>=8 -> k = kt*32 + 16 + 8g + (j-8).
__device__ __forceinline__ v16h ldsA(const f16* buf, int lane, int kt) {
  int m = lane & 15, g = (lane >> 4) & 1;
  const f16* p = buf + m * LDST + kt * 32 + 8 * g;
  V16u u;
  u.h[0] = *(const v8h*)p;
  u.h[1] = *(const v8h*)(p + 16);
  return u.v;
}

// C/D layout: n = lane&15 (+16*nb), m = r + 8g. Bias + optional ReLU -> LDS f16.
__device__ __forceinline__ void epi_lds(const v8f* c, const float* __restrict__ bias,
                                        bool relu, f16* lds, int lane) {
  int n0 = lane & 15, g = (lane >> 4) & 1;
#pragma unroll
  for (int nb = 0; nb < 4; ++nb) {
    int n = nb * 16 + n0;
    float bv = bias[n];
#pragma unroll
    for (int r = 0; r < 8; ++r) {
      float v = c[nb][r] + bv;
      if (relu) v = fmaxf(v, 0.f);
      lds[(r + 8 * g) * LDST + n] = (f16)v;
    }
  }
}

// ---------------- workspace init kernels ----------------

__global__ void zero_f32(float* __restrict__ p, int n) {
  int i = blockIdx.x * blockDim.x + threadIdx.x;
  if (i < n) p[i] = 0.f;
}

// f32 [Ksrc x Nsrc] row-major -> f16 packed B-fragment order, zero-padded.
__global__ void cvt_pack(const float* __restrict__ w, f16* __restrict__ dst,
                         int Ksrc, int Nsrc, int KT, int NB) {
  int total = NB * KT * 512;
  for (int i = blockIdx.x * blockDim.x + threadIdx.x; i < total;
       i += gridDim.x * blockDim.x) {
    int j = i & 15;
    int lane = (i >> 4) & 31;
    int t = i >> 9;          // nb*KT + kt
    int kt = t % KT, nb = t / KT;
    int k = kt * 32 + 8 * (lane >> 4) + (j & 7) + 16 * (j >> 3);
    int n = nb * 16 + (lane & 15);
    float v = (k < Ksrc && n < Nsrc) ? w[k * Nsrc + n] : 0.f;
    dst[i] = (f16)v;
  }
}

// ---------------- node encoders ----------------

__device__ __forceinline__ void encode_one(
    const float* __restrict__ x2, const f16* __restrict__ w1p,
    const f16* __restrict__ w2p, const float* __restrict__ b1,
    const float* __restrict__ b2, f16* __restrict__ enc, int colOff,
    f16* lbuf, int base, int lane) {
  int m0 = lane & 15, g = (lane >> 4) & 1;
  int node = base + m0;
  float2 xv = make_float2(0.f, 0.f);
  if (g == 0) xv = *(const float2*)(x2 + (long)node * 2);

  auto gA1 = [&](int kt) -> v16h {   // K padded 2->32: only lanes g==0, j<2 carry data
    (void)kt;
    V16u u;
#pragma unroll
    for (int j = 0; j < 16; ++j) u.e[j] = (f16)0.f;
    if (g == 0) { u.e[0] = (f16)xv.x; u.e[1] = (f16)xv.y; }
    return u.v;
  };
  v8f c[4] = {};
  mm_acc<1>(gA1, w1p, c, lane);
  epi_lds(c, b1, true, lbuf, lane);
  __threadfence_block();

  v8f d[4] = {};
  auto gA2 = [&](int kt) -> v16h { return ldsA(lbuf, lane, kt); };
  mm_acc<2>(gA2, w2p, d, lane);
#pragma unroll
  for (int nb = 0; nb < 4; ++nb) {
    int n = nb * 16 + m0;
    float bv = b2[n];
#pragma unroll
    for (int r = 0; r < 8; ++r)
      enc[(long)(base + r + 8 * g) * 128 + colOff + n] = (f16)(d[nb][r] + bv);
  }
}

__global__ void __launch_bounds__(256) k_encode(
    const float* __restrict__ pos, const float* __restrict__ vel,
    const f16* __restrict__ wpk, const float* __restrict__ b1p,
    const float* __restrict__ b2p, const float* __restrict__ b1v,
    const float* __restrict__ b2v, f16* __restrict__ enc) {
  __shared__ alignas(32) f16 smem[8][16 * LDST];
  int lane = threadIdx.x & 31, wave = threadIdx.x >> 5;
  int tile = blockIdx.x * 8 + wave;
  if (tile >= NN / 16) return;
  int base = tile * 16;
  encode_one(pos, wpk + OFF_PE_W1, wpk + OFF_PE_W2, b1p, b2p, enc, 0, smem[wave],
             base, lane);
  encode_one(vel, wpk + OFF_VE_W1, wpk + OFF_VE_W2, b1v, b2v, enc, 64, smem[wave],
             base, lane);
}

// ---------------- edge MLP + scatter-add ----------------

__global__ void __launch_bounds__(256) k_edge(
    const float* __restrict__ pos, const int* __restrict__ ei,
    const f16* __restrict__ wpk, const float* __restrict__ b1,
    const float* __restrict__ b2, float* __restrict__ agg) {
  __shared__ alignas(32) f16 smem[8][16 * LDST];
  int lane = threadIdx.x & 31, wave = threadIdx.x >> 5;
  int tile = blockIdx.x * 8 + wave;          // NE/16 = 100000 tiles, exact grid
  if (tile >= NE / 16) return;
  int e0 = tile * 16;
  int m0 = lane & 15, g = (lane >> 4) & 1;
  f16* lbuf = smem[wave];

  float4 x = make_float4(0.f, 0.f, 0.f, 0.f);
  if (g == 0) {                              // gather [pos[src], pos[dst]] for row m0
    int s = ei[e0 + m0];
    int d = ei[NE + e0 + m0];
    float2 ps = *(const float2*)(pos + (long)s * 2);
    float2 pd = *(const float2*)(pos + (long)d * 2);
    x = make_float4(ps.x, ps.y, pd.x, pd.y);
  }
  auto gA1 = [&](int kt) -> v16h {           // K padded 4->32
    (void)kt;
    V16u u;
#pragma unroll
    for (int j = 0; j < 16; ++j) u.e[j] = (f16)0.f;
    if (g == 0) {
      u.e[0] = (f16)x.x; u.e[1] = (f16)x.y; u.e[2] = (f16)x.z; u.e[3] = (f16)x.w;
    }
    return u.v;
  };
  v8f c[4] = {};
  mm_acc<1>(gA1, wpk + OFF_EE_W1, c, lane);
  epi_lds(c, b1, true, lbuf, lane);
  __threadfence_block();

  v8f d[4] = {};
  auto gA2 = [&](int kt) -> v16h { return ldsA(lbuf, lane, kt); };
  mm_acc<2>(gA2, wpk + OFF_EE_W2, d, lane);

  int dsts[8];
#pragma unroll
  for (int r = 0; r < 8; ++r) dsts[r] = ei[NE + e0 + r + 8 * g];
#pragma unroll
  for (int nb = 0; nb < 4; ++nb) {
    int n = nb * 16 + m0;
    float bv = b2[n];
#pragma unroll
    for (int r = 0; r < 8; ++r)
      atomicAdd(agg + (long)dsts[r] * 64 + n, d[nb][r] + bv);   // L2-resident atomics
  }
}

// ---------------- process MLP + decoders ----------------

__device__ __forceinline__ void dec_out(const f16* bufH, const f16* __restrict__ wpk,
                                        const float* __restrict__ b2,
                                        float* __restrict__ outp, int base, int lane) {
  int n0 = lane & 15, g = (lane >> 4) & 1;
  v8f c = {};
#pragma unroll
  for (int kt = 0; kt < 2; ++kt) {
    v16h a = ldsA(bufH, lane, kt);
    v16h b = *(const v16h*)(wpk + ((kt * 32 + lane) << 4));   // nb==0, N padded 2->16
    c = wmma32(a, b, c);
  }
  if (n0 < 2) {
    float bv = b2[n0];
#pragma unroll
    for (int r = 0; r < 8; ++r)
      outp[(base + r + 8 * g) * 2 + n0] = c[r] + bv;
  }
}

__global__ void __launch_bounds__(256) k_process(
    const f16* __restrict__ enc, const float* __restrict__ agg,
    const f16* __restrict__ wpk, const float* __restrict__ b1P,
    const float* __restrict__ b2P, const float* __restrict__ b1pd,
    const float* __restrict__ b2pd, const float* __restrict__ b1vd,
    const float* __restrict__ b2vd, float* __restrict__ out) {
  __shared__ alignas(32) f16 smem[8][2][16 * LDST];
  int lane = threadIdx.x & 31, wave = threadIdx.x >> 5;
  int tile = blockIdx.x * 8 + wave;
  if (tile >= NN / 16) return;
  int base = tile * 16;
  int m0 = lane & 15, g = (lane >> 4) & 1;
  int node = base + m0;
  f16* bufA = smem[wave][0];
  f16* bufB = smem[wave][1];
  const f16* encRow = enc + (long)node * 128;
  const float* aggRow = agg + (long)node * 64;

  // A input [16 x 192]: cols 0-127 = enc (f16), cols 128-191 = agg (f32 -> f16)
  auto gA = [&](int kt) -> v16h {
    V16u u;
    if (kt < 4) {
      const f16* p = encRow + kt * 32 + 8 * g;
      u.h[0] = *(const v8h*)p;
      u.h[1] = *(const v8h*)(p + 16);
    } else {
      const float* q = aggRow + (kt - 4) * 32 + 8 * g;
#pragma unroll
      for (int j = 0; j < 8; ++j) { u.e[j] = (f16)q[j]; u.e[8 + j] = (f16)q[16 + j]; }
    }
    return u.v;
  };
  v8f c[4] = {};
  mm_acc<6>(gA, wpk + OFF_PR_W1, c, lane);
  epi_lds(c, b1P, true, bufB, lane);
  __threadfence_block();

  v8f h[4] = {};
  auto gB = [&](int kt) -> v16h { return ldsA(bufB, lane, kt); };
  mm_acc<2>(gB, wpk + OFF_PR_W2, h, lane);
  epi_lds(h, b2P, false, bufA, lane);       // node_hidden (no ReLU on output)
  __threadfence_block();

  auto gH = [&](int kt) -> v16h { return ldsA(bufA, lane, kt); };

  v8f p1[4] = {};
  mm_acc<2>(gH, wpk + OFF_PD_W1, p1, lane);
  epi_lds(p1, b1pd, true, bufB, lane);
  __threadfence_block();
  dec_out(bufB, wpk + OFF_PD_W2, b2pd, out, base, lane);            // pos_hat

  v8f v1[4] = {};
  mm_acc<2>(gH, wpk + OFF_VD_W1, v1, lane);
  epi_lds(v1, b1vd, true, bufB, lane);
  __threadfence_block();
  dec_out(bufB, wpk + OFF_VD_W2, b2vd, out + 2 * NN, base, lane);   // vel_hat
}

// ---------------- host launcher ----------------

extern "C" void kernel_launch(void* const* d_in, const int* in_sizes, int n_in,
                              void* d_out, int out_size, void* d_ws, size_t ws_size,
                              hipStream_t stream) {
  (void)in_sizes; (void)n_in; (void)out_size; (void)ws_size;
  const float* pos = (const float*)d_in[0];
  const float* vel = (const float*)d_in[1];
  const int*   ei  = (const int*)d_in[2];
  // params in setup_inputs() dict order: pos_enc, vel_enc, edge_enc, process,
  // pos_dec, vel_dec; each MLP as w1, b1, w2, b2.
  const float* pe_w1 = (const float*)d_in[3];
  const float* pe_b1 = (const float*)d_in[4];
  const float* pe_w2 = (const float*)d_in[5];
  const float* pe_b2 = (const float*)d_in[6];
  const float* ve_w1 = (const float*)d_in[7];
  const float* ve_b1 = (const float*)d_in[8];
  const float* ve_w2 = (const float*)d_in[9];
  const float* ve_b2 = (const float*)d_in[10];
  const float* ee_w1 = (const float*)d_in[11];
  const float* ee_b1 = (const float*)d_in[12];
  const float* ee_w2 = (const float*)d_in[13];
  const float* ee_b2 = (const float*)d_in[14];
  const float* pr_w1 = (const float*)d_in[15];
  const float* pr_b1 = (const float*)d_in[16];
  const float* pr_w2 = (const float*)d_in[17];
  const float* pr_b2 = (const float*)d_in[18];
  const float* pd_w1 = (const float*)d_in[19];
  const float* pd_b1 = (const float*)d_in[20];
  const float* pd_w2 = (const float*)d_in[21];
  const float* pd_b2 = (const float*)d_in[22];
  const float* vd_w1 = (const float*)d_in[23];
  const float* vd_b1 = (const float*)d_in[24];
  const float* vd_w2 = (const float*)d_in[25];
  const float* vd_b2 = (const float*)d_in[26];
  float* out = (float*)d_out;

  // Workspace: agg f32 [N*64] | enc f16 [N*128] | packed f16 weights (~90KB)
  char* ws = (char*)d_ws;
  float* agg = (float*)ws;
  f16* enc = (f16*)(ws + (size_t)NN * 64 * 4);
  f16* wpk = (f16*)(ws + (size_t)NN * 64 * 4 + (size_t)NN * 128 * 2);

  zero_f32<<<(NN * 64 + 255) / 256, 256, 0, stream>>>(agg, NN * 64);

  auto pack = [&](const float* w, int off, int Ks, int Ns, int KT, int NB) {
    int total = NB * KT * 512;
    cvt_pack<<<(total + 255) / 256, 256, 0, stream>>>(w, wpk + off, Ks, Ns, KT, NB);
  };
  pack(pe_w1, OFF_PE_W1, 2, 64, 1, 4);
  pack(pe_w2, OFF_PE_W2, 64, 64, 2, 4);
  pack(ve_w1, OFF_VE_W1, 2, 64, 1, 4);
  pack(ve_w2, OFF_VE_W2, 64, 64, 2, 4);
  pack(ee_w1, OFF_EE_W1, 4, 64, 1, 4);
  pack(ee_w2, OFF_EE_W2, 64, 64, 2, 4);
  pack(pr_w1, OFF_PR_W1, 192, 64, 6, 4);
  pack(pr_w2, OFF_PR_W2, 64, 64, 2, 4);
  pack(pd_w1, OFF_PD_W1, 64, 64, 2, 4);
  pack(pd_w2, OFF_PD_W2, 64, 2, 2, 1);
  pack(vd_w1, OFF_VD_W1, 64, 64, 2, 4);
  pack(vd_w2, OFF_VD_W2, 64, 2, 2, 1);

  k_encode<<<(NN / 16 + 7) / 8, 256, 0, stream>>>(pos, vel, wpk, pe_b1, pe_b2,
                                                  ve_b1, ve_b2, enc);
  k_edge<<<(NE / 16) / 8, 256, 0, stream>>>(pos, ei, wpk, ee_b1, ee_b2, agg);
  k_process<<<(NN / 16 + 7) / 8, 256, 0, stream>>>(enc, agg, wpk, pr_b1, pr_b2,
                                                   pd_b1, pd_b2, vd_b1, vd_b2, out);
}
